// GPUOptimizedMamba2Block_27685359190818
// MI455X (gfx1250) — compile-verified, matrix operand order checked by reading, not run on hardware
//
#include <hip/hip_runtime.h>
#include <hip/hip_bf16.h>
#include <math.h>

typedef __attribute__((ext_vector_type(16))) __bf16 v16bf;
typedef __attribute__((ext_vector_type(8)))  __bf16 v8bf;
typedef __attribute__((ext_vector_type(2)))  __bf16 v2bf;
typedef __attribute__((ext_vector_type(8)))  float  v8f;

#define BLK_M 128
#define BLK_N 128
#define BLK_K 32
#define LDA   40   // bf16 elems per A row in LDS (80B stride, 16B-aligned frags)
#define LDB   40   // bf16 elems per Bt row in LDS

union Frag16 { v16bf v; v8bf h[2]; };

__device__ inline v16bf load_frag(const __bf16* p) {
    Frag16 f;
    f.h[0] = *(const v8bf*)(p);       // ds_load_b128
    f.h[1] = *(const v8bf*)(p + 8);   // ds_load_b128
    return f.v;
}

// Single v_cvt_pk_bf16_f32: convert+pack two floats, one 32-bit LDS store.
__device__ inline void store_bf16x2(__bf16* dst, float a, float b) {
    v2bf p;
    p[0] = (__bf16)a;
    p[1] = (__bf16)b;
    *(v2bf*)dst = p;
}

// C[M,N] = A[M,K] @ B[K,N]; fp32 in/out, bf16 WMMA, f32 accumulate.
// M % 128 == 0, K % 32 == 0 assumed. NGUARD handles N % 128 != 0.
// SOFTPLUS applies softplus(v + bias[n]) in the epilogue.
// 8 waves = 4 (M) x 2 (N); wave tile 32x64 -> 8 WMMAs / K-step / wave.
template <bool NGUARD, bool SOFTPLUS>
__global__ __launch_bounds__(256) void wmma_gemm(
    const float* __restrict__ A, const float* __restrict__ B,
    float* __restrict__ C, const float* __restrict__ bias,
    int M, int N, int K)
{
    __shared__ __bf16 As[2][BLK_M * LDA];   // double-buffered A tile
    __shared__ __bf16 Bt[2][BLK_N * LDB];   // double-buffered B tile (transposed)

    const int tid  = threadIdx.x;
    const int lane = tid & 31;
    const int w    = tid >> 5;       // 0..7
    const int wm   = w >> 1;         // 4 waves along M (32 rows each)
    const int wn   = w & 1;          // 2 waves along N (64 cols each)
    const int row0 = blockIdx.x * BLK_M;
    const int col0 = blockIdx.y * BLK_N;

    const int r16 = lane & 15;
    const int kh  = lane >> 4;

    v8f acc[2][4] = {};

    // Register staging (pipeline stage between global and LDS)
    float2 aR[8];
    float  bR0[8], bR1[8];

    auto issue_loads = [&](int k0) {
        #pragma unroll
        for (int j = 0; j < 8; ++j) {          // A: 128x32 = 2048 float2
            const int p = tid + j * 256;
            const int r = p >> 4, cp = p & 15;
            aR[j] = *(const float2*)&A[(size_t)(row0 + r) * K + k0 + 2 * cp];
        }
        #pragma unroll
        for (int j = 0; j < 8; ++j) {          // B: 32x128, k-pairs per column
            const int p   = tid + j * 256;
            const int col = p & 127, kp = p >> 7;
            const int gc  = col0 + col;
            if (!NGUARD || gc < N) {
                const float* g = &B[(size_t)(k0 + 2 * kp) * N + gc];
                bR0[j] = g[0];
                bR1[j] = g[N];
            } else {
                bR0[j] = 0.f;
                bR1[j] = 0.f;
            }
        }
    };
    auto commit_tiles = [&](int pb) {
        #pragma unroll
        for (int j = 0; j < 8; ++j) {
            const int p = tid + j * 256;
            const int r = p >> 4, cp = p & 15;
            store_bf16x2(&As[pb][r * LDA + 2 * cp], aR[j].x, aR[j].y);
        }
        #pragma unroll
        for (int j = 0; j < 8; ++j) {
            const int p   = tid + j * 256;
            const int col = p & 127, kp = p >> 7;
            store_bf16x2(&Bt[pb][col * LDB + 2 * kp], bR0[j], bR1[j]);
        }
    };

    // Pipeline prologue: tile 0 -> LDS buf 0, tile 1 in flight to registers
    issue_loads(0);
    commit_tiles(0);
    if (BLK_K < K) issue_loads(BLK_K);

    int pb = 0;
    for (int k0 = 0; k0 < K; k0 += BLK_K, pb ^= 1) {
        __syncthreads();   // single barrier per K-step (ping-pong buffers)

        v16bf af[2], bf[4];
        #pragma unroll
        for (int r = 0; r < 2; ++r)
            af[r] = load_frag(&As[pb][(wm * 32 + r * 16 + r16) * LDA + kh * 16]);
        #pragma unroll
        for (int q = 0; q < 4; ++q)
            bf[q] = load_frag(&Bt[pb][(wn * 64 + q * 16 + r16) * LDB + kh * 16]);

        #pragma unroll
        for (int r = 0; r < 2; ++r)
            #pragma unroll
            for (int q = 0; q < 4; ++q)
                acc[r][q] = __builtin_amdgcn_wmma_f32_16x16x32_bf16(
                    false, af[r], false, bf[q], (short)0, acc[r][q], false, false);

        if (k0 + BLK_K < K) {
            commit_tiles(pb ^ 1);                      // fill other buffer
            if (k0 + 2 * BLK_K < K)
                issue_loads(k0 + 2 * BLK_K);           // keep loads in flight
        }
    }

    // D layout: VGPR r8 -> M = r8 + 8*kh within 16-row tile; N = lane&15
    auto epi = [&](float v, int n) -> float {
        if (SOFTPLUS) {
            v += bias[n];
            v = (v > 20.f) ? v : log1pf(__expf(v));
        }
        return v;
    };
    #pragma unroll
    for (int r = 0; r < 2; ++r) {
        const int mb = row0 + wm * 32 + r * 16 + kh * 8;
        #pragma unroll
        for (int q = 0; q < 4; ++q) {
            const int nb = col0 + wn * 64 + q * 16 + r16;
            if (!NGUARD || nb < N) {
                #pragma unroll
                for (int r8 = 0; r8 < 8; ++r8)
                    C[(size_t)(mb + r8) * N + nb] = epi(acc[r][q][r8], nb);
            }
        }
    }
}

// Causal depthwise conv1d (width 4) + bias + SiLU over the x_ssm half of xz.
__global__ void conv_silu_kernel(const float* __restrict__ xz,
                                 const float* __restrict__ cw,
                                 const float* __restrict__ cb,
                                 float* __restrict__ u,
                                 int B, int L, int D)
{
    const int idx = blockIdx.x * blockDim.x + threadIdx.x;
    const int total = B * L * D;
    if (idx >= total) return;
    const int c = idx % D;
    const int r = idx / D;            // b*L + t
    const int t = r % L;
    const int s2 = 2 * D;
    float acc = cb[c];
    #pragma unroll
    for (int k = 0; k < 4; ++k) {
        const int tt = t - 3 + k;
        if (tt >= 0)
            acc += xz[(size_t)(r - 3 + k) * s2 + c] * cw[c * 4 + k];
    }
    u[idx] = acc / (1.f + __expf(-acc));   // SiLU
}

// Selective scan: one 16-lane group per (b,d) pair, lane = state index n.
__global__ __launch_bounds__(256) void scan_kernel(
    const float* __restrict__ u,  const float* __restrict__ dt,
    const float* __restrict__ bc, const float* __restrict__ xz,
    const float* __restrict__ A_log, const float* __restrict__ Dv,
    float* __restrict__ y, int B, int L, int D, int N)
{
    const int pair   = blockIdx.x * (blockDim.x >> 4) + (threadIdx.x >> 4);
    const int lane_n = threadIdx.x & 15;
    if (pair >= B * D) return;
    const int b   = pair / D;
    const int dch = pair % D;

    const float Aval = -__expf(A_log[dch * N + lane_n]);
    const float Dd   = Dv[dch];
    float s = 0.f;

    const float* up  = u  + (size_t)b * L * D + dch;
    const float* dtp = dt + (size_t)b * L * D + dch;
    const float* bp  = bc + (size_t)b * L * 2 * N + lane_n;
    const float* zp  = xz + (size_t)b * L * 2 * D + D + dch;
    float*       yp  = y  + (size_t)b * L * D + dch;

    #pragma unroll 4
    for (int t = 0; t < L; ++t) {
        const float ut  = up[(size_t)t * D];
        const float dtt = dtp[(size_t)t * D];
        const float Bt  = bp[(size_t)t * 2 * N];
        const float Ct  = bp[(size_t)t * 2 * N + N];

        s = s * __expf(dtt * Aval) + Bt * ut;
        float p = s * Ct;
        p += __shfl_xor(p, 8, 32);
        p += __shfl_xor(p, 4, 32);
        p += __shfl_xor(p, 2, 32);
        p += __shfl_xor(p, 1, 32);

        if (lane_n == 0) {
            const float z = zp[(size_t)t * 2 * D];
            yp[(size_t)t * D] = (p + Dd * ut) * (z / (1.f + __expf(-z)));
        }
    }
}

extern "C" void kernel_launch(void* const* d_in, const int* in_sizes, int n_in,
                              void* d_out, int out_size, void* d_ws, size_t ws_size,
                              hipStream_t stream)
{
    (void)in_sizes; (void)n_in; (void)out_size; (void)ws_size;

    const float* x      = (const float*)d_in[0];
    const float* state  = (const float*)d_in[1];
    const float* W_in   = (const float*)d_in[2];
    const float* conv_w = (const float*)d_in[3];
    const float* conv_b = (const float*)d_in[4];
    const float* W_x    = (const float*)d_in[5];
    const float* W_dt   = (const float*)d_in[6];
    const float* b_dt   = (const float*)d_in[7];
    const float* A_log  = (const float*)d_in[8];
    const float* Dvec   = (const float*)d_in[9];
    const float* W_out  = (const float*)d_in[10];
    float* out = (float*)d_out;

    const int B = 4, L = 4096, D = 768, N = 16;
    const int M = B * L;                       // 16384

    float* ws      = (float*)d_ws;
    float* buf_xz  = ws;                               // M * 1536
    float* buf_u   = buf_xz + (size_t)M * 2 * D;       // M * 768
    float* buf_dt  = buf_u  + (size_t)M * D;           // M * 768
    float* buf_bc  = buf_dt + (size_t)M * D;           // M * 32
    float* buf_y   = buf_bc + (size_t)M * 2 * N;       // M * 768

    const dim3 blk(256);

    // 1) xz = X @ W_in   [16384 x 1536]
    { dim3 g(M / BLK_M, (2 * D) / BLK_N);
      wmma_gemm<false, false><<<g, blk, 0, stream>>>(
          x, W_in, buf_xz, nullptr, M, 2 * D, D); }

    // 2) causal conv + SiLU -> u
    { const int total = M * D;
      conv_silu_kernel<<<(total + 255) / 256, blk, 0, stream>>>(
          buf_xz, conv_w, conv_b, buf_u, B, L, D); }

    // 3+4) dt = softplus(u @ W_dt + b_dt), fused epilogue
    { dim3 g(M / BLK_M, D / BLK_N);
      wmma_gemm<false, true><<<g, blk, 0, stream>>>(
          buf_u, W_dt, buf_dt, b_dt, M, D, D); }

    // 5) (B,C) = u @ W_x   [16384 x 32] -> guarded (N < 128)
    { dim3 g(M / BLK_M, 1);
      wmma_gemm<true, false><<<g, blk, 0, stream>>>(
          buf_u, W_x, buf_bc, nullptr, M, 2 * N, D); }

    // 6) selective scan + D-skip + SiLU(z) gating -> y
    scan_kernel<<<(B * D) / 16, blk, 0, stream>>>(
        buf_u, buf_dt, buf_bc, buf_xz, A_log, Dvec, buf_y, B, L, D, N);

    // 7) out = y @ W_out   [16384 x 768]
    { dim3 g(M / BLK_M, D / BLK_N);
      wmma_gemm<false, false><<<g, blk, 0, stream>>>(
          buf_y, W_out, out, nullptr, M, D, D); }

    // 8) reference returns the incoming state unchanged as new_state
    (void)hipMemcpyAsync(out + (size_t)M * D, state,
                         sizeof(float) * (size_t)B * D * N,
                         hipMemcpyDeviceToDevice, stream);
}
